// CIN_15290083573767
// MI455X (gfx1250) — compile-verified, matrix operand order checked by reading, not run on hardware
//
#include <hip/hip_runtime.h>
#include <stdint.h>
#include <stddef.h>

// ---------------- CDNA5 WMMA types ----------------
typedef __attribute__((ext_vector_type(16))) _Float16 v16h;
typedef __attribute__((ext_vector_type(8)))  float    v8f;
typedef __attribute__((ext_vector_type(2)))  _Float16 h2;
typedef __attribute__((ext_vector_type(4)))  int      v4i;

// ---------------- problem constants ----------------
#define BATCH   2048
#define F_ORIG  39
#define F_PAD   40      // even pad: WMMA K-pairs never straddle an h-row
#define E_DIM   64
#define O_DIM   64
#define KPAD0   (40 * F_PAD)   // layer 0: Hk=39 padded to 40 -> 1600
#define KPAD12  (64 * F_PAD)   // layers 1,2: 2560

// workspace layout (bytes, 256-aligned)
static constexpr size_t OFF_W0  = 0;          // 1600*64*2  = 204800
static constexpr size_t OFF_W1  = 204800;     // 2560*64*2  = 327680
static constexpr size_t OFF_W2  = 532480;     // 327680
static constexpr size_t OFF_X0B = 860160;     // 2048*64*40*2 = 10485760
static constexpr size_t OFF_X1  = 11345920;   // 2048*64*64*2 = 16777216
static constexpr size_t OFF_X2  = 28123136;
static constexpr size_t OFF_X3  = 44900352;   // end ~ 61.7 MB

union ABfrag { unsigned int u[8]; h2 h[8]; v16h v; };

// magic div by 40, exact for 0 <= k < 2560
__device__ __forceinline__ int div40(int k) {
  return (int)(((unsigned)k * 26215u) >> 20);
}

// ---------------- gfx1250 async global->LDS copy (ASYNCcnt path) ----------------
#if defined(__gfx1250__) && __has_builtin(__builtin_amdgcn_global_load_async_to_lds_b128)
#define HAVE_ASYNC_LDS 1
__device__ __forceinline__ void async_cp16(const void* g, void* l) {
  __builtin_amdgcn_global_load_async_to_lds_b128(
      (__attribute__((address_space(1))) v4i*)(uintptr_t)g,
      (__attribute__((address_space(3))) v4i*)(unsigned int)(uintptr_t)l,
      0, 0);
}
__device__ __forceinline__ void async_wait0() {
#if __has_builtin(__builtin_amdgcn_s_wait_asynccnt)
  __builtin_amdgcn_s_wait_asynccnt(0);
#else
  asm volatile("s_wait_asynccnt 0" ::: "memory");
#endif
}
#else
#define HAVE_ASYNC_LDS 0
#endif

// ---------------------------------------------------------------
// Prep: W[o][h*39+f] fp32 -> fp16 fragments in B-operand lane order.
// frag half index: idx = (((s*4 + nt)*32 + L)*16 + j)
//   lane L: col n = L&15, kg = L>>4 ; half j: koff = (j&8?16:0)+kg*8+(j&7)
// ---------------------------------------------------------------
__global__ void prep_w(const float* __restrict__ W, _Float16* __restrict__ Wf,
                       int Hk, int total) {
  int idx = blockIdx.x * blockDim.x + threadIdx.x;
  if (idx >= total) return;
  int j  = idx & 15;
  int L  = (idx >> 4) & 31;
  int nt = (idx >> 9) & 3;
  int s  = idx >> 11;
  int n  = L & 15, kg = L >> 4;
  int koff = ((j & 8) ? 16 : 0) + kg * 8 + (j & 7);
  int k = 32 * s + koff;
  int h = div40(k), f = k - F_PAD * h;
  int o = nt * 16 + n;
  float v = 0.0f;
  if (h < Hk && f < F_ORIG) v = W[(size_t)o * (Hk * F_ORIG) + h * F_ORIG + f];
  Wf[idx] = (_Float16)v;
}

// X[b][f][e] fp32 -> X0b[b][e][f(40)] fp16, padded f=39 row zeroed
__global__ void prep_x0(const float* __restrict__ X, _Float16* __restrict__ X0b) {
  int idx = blockIdx.x * blockDim.x + threadIdx.x;   // total = 2048*64*40 (exact grid)
  int f = idx % F_PAD;
  int e = (idx / F_PAD) & (E_DIM - 1);
  int b = idx / (F_PAD * E_DIM);
  float v = (f < F_ORIG) ? X[((size_t)b * F_ORIG + f) * E_DIM + e] : 0.0f;
  X0b[idx] = (_Float16)v;
}

// ---------------------------------------------------------------
// CIN layer as on-the-fly GEMM:
//   out[(b,e), o] = sum_k  A[(b,e),k] * Wf[k,o],  A = Xk[h,e]*X0[f,e]
// Block: 256 threads = 8 wave32, 4 batches, M=256 rows (b,e),
// each wave: 2 M-tiles x 4 N-tiles, fp32 accum, 8 WMMAs / K-step.
// ---------------------------------------------------------------
template <bool USE_XK>
__global__ __launch_bounds__(256)
void cin_layer(const _Float16* __restrict__ x0b,   // [B][64][40] fp16
               const _Float16* __restrict__ xkb,   // [B][64][64] fp16 (USE_XK)
               const _Float16* __restrict__ wfrag, // B-fragment order
               const float*    __restrict__ bias,  // [64]
               _Float16*       __restrict__ xout,  // [B][64][64] fp16
               int Kpad) {
  __shared__ __align__(16) _Float16 x0s[4 * E_DIM * F_PAD];  // 20 KB
  __shared__ __align__(16) _Float16 xks[4 * E_DIM * E_DIM];  // 32 KB

  const int tid = threadIdx.x;

  // ---- stage this block's 4 batches into LDS (fp16, transposed e-major) ----
  {
    const uint4* s0 = (const uint4*)(x0b + (size_t)blockIdx.x * 4 * E_DIM * F_PAD);
    uint4* d0 = (uint4*)x0s;
    const uint4* s1 = (const uint4*)(xkb + (size_t)blockIdx.x * 4 * E_DIM * E_DIM);
    uint4* d1 = (uint4*)xks;
#if HAVE_ASYNC_LDS
#pragma unroll
    for (int i = 0; i < 5; ++i) async_cp16(s0 + tid + 256 * i, d0 + tid + 256 * i);
    if (USE_XK) {
#pragma unroll
      for (int i = 0; i < 8; ++i) async_cp16(s1 + tid + 256 * i, d1 + tid + 256 * i);
    }
    async_wait0();
#else
#pragma unroll
    for (int i = 0; i < 5; ++i) d0[tid + 256 * i] = s0[tid + 256 * i];   // 1280 x 16B
    if (USE_XK) {
#pragma unroll
      for (int i = 0; i < 8; ++i) d1[tid + 256 * i] = s1[tid + 256 * i]; // 2048 x 16B
    }
#endif
  }
  __syncthreads();

  const int L = tid & 31, wid = tid >> 5;
  const int lane16 = L & 15, kg = L >> 4;

  int gI[2], rowBase[2];
#pragma unroll
  for (int mt = 0; mt < 2; ++mt) {
    int t = 2 * wid + mt;          // m-tile 0..15
    gI[mt] = t >> 2;               // batch within block
    int e = ((t & 3) << 4) + lane16;
    rowBase[mt] = gI[mt] * E_DIM + e;
  }

  v8f C[2][4];
#pragma unroll
  for (int mt = 0; mt < 2; ++mt)
#pragma unroll
    for (int nt = 0; nt < 4; ++nt) C[mt][nt] = (v8f){0.f,0.f,0.f,0.f,0.f,0.f,0.f,0.f};

  const uint4* wbase = (const uint4*)wfrag;
  const int nIter = Kpad >> 5;

  for (int s = 0; s < nIter; ++s) {
    const int k0 = s << 5;

    // B fragments: two coalesced 16B loads per lane per N-tile (L2-resident W)
    ABfrag bfr[4];
#pragma unroll
    for (int nt = 0; nt < 4; ++nt) {
      const uint4* p = wbase + ((((size_t)(s * 4 + nt)) * 32 + L) << 1);
      ((uint4*)bfr[nt].u)[0] = p[0];
      ((uint4*)bfr[nt].u)[1] = p[1];
    }
    if (s + 1 < nIter)   // pull next W slab toward the WGP (global_prefetch_b8)
      __builtin_prefetch(wbase + ((((size_t)((s + 1) * 4)) * 32 + L) << 1), 0, 1);

    // Build BOTH A fragments first: two live A buffers -> the pk_muls of the
    // next iteration land >8 WMMAs after the last read of these registers,
    // covering the WMMA_F16 -> VALU WAR hazard without v_nops.
    ABfrag a[2];
#pragma unroll
    for (int mt = 0; mt < 2; ++mt) {
      const _Float16* hrow = USE_XK ? &xks[rowBase[mt] * E_DIM]
                                    : &x0s[rowBase[mt] * F_PAD];
      const _Float16* frow = &x0s[rowBase[mt] * F_PAD];
#pragma unroll
      for (int p = 0; p < 8; ++p) {
        // A-layout: VGPR p holds K = (p&4?16:0) + kg*8 + 2*(p&3), +1
        int koff = ((p & 4) ? 16 : 0) + kg * 8 + ((p & 3) << 1);
        int k = k0 + koff;                       // even -> f even -> aligned pair
        int h = div40(k);
        int f = k - F_PAD * h;
        _Float16 xk = hrow[h];
        h2 xk2 = {xk, xk};
        h2 x0p = *(const h2*)&frow[f];
        a[mt].h[p] = xk2 * x0p;                  // v_pk_mul_f16
      }
    }

#pragma unroll
    for (int mt = 0; mt < 2; ++mt)
#pragma unroll
      for (int nt = 0; nt < 4; ++nt)
        C[mt][nt] = __builtin_amdgcn_wmma_f32_16x16x32_f16(
            false, a[mt].v, false, bfr[nt].v, (short)0, C[mt][nt], false, false);
  }

  // ---- epilogue: bias + store fp16 as [b][e][o] (next layer's A-h operand) ----
#pragma unroll
  for (int mt = 0; mt < 2; ++mt) {
    int t = 2 * wid + mt;
    int ebase = (t & 3) << 4;
    size_t bglob = (size_t)blockIdx.x * 4 + gI[mt];
#pragma unroll
    for (int nt = 0; nt < 4; ++nt) {
      int o = nt * 16 + lane16;
      float bv = bias[o];
#pragma unroll
      for (int r = 0; r < 8; ++r) {
        int e = ebase + r + kg * 8;              // C: VGPR r = rows r / r+8
        xout[(bglob * E_DIM + e) * O_DIM + o] = (_Float16)(C[mt][nt][r] + bv);
      }
    }
  }
}

// ---------------------------------------------------------------
// Final: out[b] = fc_b + sum_{l,o} fc_w[l*64+o] * sum_e Xl[b][e][o]
// ---------------------------------------------------------------
__global__ __launch_bounds__(256)
void cin_final(const _Float16* __restrict__ X1, const _Float16* __restrict__ X2,
               const _Float16* __restrict__ X3,
               const float* __restrict__ fcw, const float* __restrict__ fcb,
               float* __restrict__ out) {
  __shared__ float red[256];
  const int b = blockIdx.x, tid = threadIdx.x;
  const _Float16* Xs[3] = {X1, X2, X3};
  float acc = 0.0f;
#pragma unroll
  for (int l = 0; l < 3; ++l) {
    const _Float16* Xb = Xs[l] + (size_t)b * (E_DIM * O_DIM);
    for (int i = tid; i < E_DIM * O_DIM; i += 256)
      acc += (float)Xb[i] * fcw[l * 64 + (i & 63)];
  }
  red[tid] = acc;
  __syncthreads();
  for (int st = 128; st > 0; st >>= 1) {
    if (tid < st) red[tid] += red[tid + st];
    __syncthreads();
  }
  if (tid == 0) out[b] = red[0] + fcb[0];
}

// ---------------------------------------------------------------
extern "C" void kernel_launch(void* const* d_in, const int* in_sizes, int n_in,
                              void* d_out, int out_size, void* d_ws, size_t ws_size,
                              hipStream_t stream) {
  (void)in_sizes; (void)n_in; (void)out_size; (void)ws_size;
  const float* X   = (const float*)d_in[0];
  const float* W0  = (const float*)d_in[1];
  const float* b0  = (const float*)d_in[2];
  const float* W1  = (const float*)d_in[3];
  const float* b1  = (const float*)d_in[4];
  const float* W2  = (const float*)d_in[5];
  const float* b2  = (const float*)d_in[6];
  const float* fcw = (const float*)d_in[7];
  const float* fcb = (const float*)d_in[8];

  char* ws = (char*)d_ws;
  _Float16* W0f = (_Float16*)(ws + OFF_W0);
  _Float16* W1f = (_Float16*)(ws + OFF_W1);
  _Float16* W2f = (_Float16*)(ws + OFF_W2);
  _Float16* X0b = (_Float16*)(ws + OFF_X0B);
  _Float16* X1b = (_Float16*)(ws + OFF_X1);
  _Float16* X2b = (_Float16*)(ws + OFF_X2);
  _Float16* X3b = (_Float16*)(ws + OFF_X3);

  prep_w<<<(KPAD0  * O_DIM + 255) / 256, 256, 0, stream>>>(W0, W0f, 39, KPAD0  * O_DIM);
  prep_w<<<(KPAD12 * O_DIM + 255) / 256, 256, 0, stream>>>(W1, W1f, 64, KPAD12 * O_DIM);
  prep_w<<<(KPAD12 * O_DIM + 255) / 256, 256, 0, stream>>>(W2, W2f, 64, KPAD12 * O_DIM);
  prep_x0<<<(BATCH * E_DIM * F_PAD) / 256, 256, 0, stream>>>(X, X0b);

  cin_layer<false><<<BATCH / 4, 256, 0, stream>>>(X0b, nullptr, W0f, b0, X1b, KPAD0);
  cin_layer<true ><<<BATCH / 4, 256, 0, stream>>>(X0b, X1b,    W1f, b1, X2b, KPAD12);
  cin_layer<true ><<<BATCH / 4, 256, 0, stream>>>(X0b, X2b,    W2f, b2, X3b, KPAD12);

  cin_final<<<BATCH, 256, 0, stream>>>(X1b, X2b, X3b, fcw, fcb, (float*)d_out);
}
